// GnnCritic_84739704750476
// MI455X (gfx1250) — matrix operand order repaired and checked by reading, not compiled, not run on hardware
//
#include <hip/hip_runtime.h>

// ---------------------------------------------------------------------------
// GNN critic, fused bf16-WMMA implementation for gfx1250 (MI455X).
//   q_t = relu( sum_o relu( relu(x_o @ W1 + b1) @ W2 + b2 ) @ RW1 + rb1 ) @ rw2 + rb2
// Dims: K1=157->160, HID=256, N2=471->480, rho hid=256. BATCH=65536, OBJ=3, towers=2.
// Compute-bound (~158 GFLOP vs ~117 MB HBM): bf16 WMMA w/ f32 accumulation,
// all intermediates LDS-resident, weights packed once to fragment layout (L2-hot).
// ---------------------------------------------------------------------------

typedef __attribute__((ext_vector_type(16))) __bf16 v16bf;
typedef __attribute__((ext_vector_type(8)))  float  v8f;

#define BATCH_N 65536
#define OBJ 3

// ws layout (bytes)
#define OFF_W1PK   0          // 2 towers * 16nt*5kt*512  bf16 = 163840 B
#define OFF_W2PK   163840     // 2 towers * 30nt*8kt*512  bf16 = 491520 B
#define OFF_RW1PK  655360     // 2 towers * 16nt*15kt*512 bf16 = 491520 B
#define OFF_BIAS   1146880    // f32: b1[2][256] b2[2][480] rb1[2][256] rw2[2][256] rb2[2]

#define N_W1PK  81920L        // ushort elems
#define N_W2PK  245760L
#define N_RW1PK 245760L
#define N_BIAS  2498L

__device__ __forceinline__ unsigned short f2bf(float f) {
    unsigned int u = __float_as_uint(f);
    u += 0x7FFFu + ((u >> 16) & 1u);            // round-to-nearest-even
    return (unsigned short)(u >> 16);
}
__device__ __forceinline__ float bf2f(unsigned short h) {
    return __uint_as_float(((unsigned int)h) << 16);
}

union Frag {
    v16bf v;
    uint4 u[2];
};

// A-fragment (16x32 bf16) from LDS, per ISA layout:
//   lane L: row M = L%16; VGPR0-3 hold K = {0..7}+(L<16?0:8), VGPR4-7 hold K = {16..23}+(L<16?0:8)
// Both half-K blocks are 8 consecutive bf16 -> one 16B ds_load_b128 each.
__device__ __forceinline__ Frag loadA(const unsigned short* s, int pitch,
                                      int m0, int k0, int lane) {
    Frag f;
    const int m  = m0 + (lane & 15);
    const int kh = (lane < 16) ? 0 : 8;
    f.u[0] = *reinterpret_cast<const uint4*>(s + m * pitch + k0 + kh);
    f.u[1] = *reinterpret_cast<const uint4*>(s + m * pitch + k0 + 16 + kh);
    return f;
}

// B-fragment (32x16 bf16) from global, pre-swizzled by the prep kernel so each
// lane reads its 16 bf16 values contiguously (2x global_load_b128).
__device__ __forceinline__ Frag loadB(const unsigned short* g, int KT,
                                      int nt, int kt, int lane) {
    Frag f;
    const uint4* p = reinterpret_cast<const uint4*>(g + ((((long)nt * KT + kt) * 32 + lane) << 4));
    f.u[0] = p[0];
    f.u[1] = p[1];
    return f;
}

// ---------------------------------------------------------------------------
// Prep kernel: bf16-convert + fragment-swizzle all weights into ws, pad with 0.
// Packed element p = ((nt*KT + kt)*32 + lane)*16 + j maps to
//   n = nt*16 + (lane&15),  k = kt*32 + (lane<16 ? 0 : 8) + (j&7) + ((j>>3)<<4)
// ---------------------------------------------------------------------------
__global__ __launch_bounds__(256)
void gnn_prep(const float* __restrict__ p1w1, const float* __restrict__ p2w1,
              const float* __restrict__ p1w2, const float* __restrict__ p2w2,
              const float* __restrict__ r1w1, const float* __restrict__ r2w1,
              const float* __restrict__ p1b1, const float* __restrict__ p2b1,
              const float* __restrict__ p1b2, const float* __restrict__ p2b2,
              const float* __restrict__ r1b1, const float* __restrict__ r2b1,
              const float* __restrict__ r1w2, const float* __restrict__ r2w2,
              const float* __restrict__ r1b2, const float* __restrict__ r2b2,
              unsigned char* __restrict__ ws)
{
    const long gid = (long)blockIdx.x * 256 + threadIdx.x;
    unsigned short* W1pk  = (unsigned short*)(ws + OFF_W1PK);
    unsigned short* W2pk  = (unsigned short*)(ws + OFF_W2PK);
    unsigned short* RW1pk = (unsigned short*)(ws + OFF_RW1PK);
    float*          bias  = (float*)(ws + OFF_BIAS);

    if (gid < N_W1PK) {                                  // phi W1: (157,256), KT=5
        const int t = (int)(gid / 40960), p = (int)(gid % 40960);
        const int j = p & 15, lane = (p >> 4) & 31, rem = p >> 9;
        const int kt = rem % 5, nt = rem / 5;
        const int n = nt * 16 + (lane & 15);
        const int k = kt * 32 + ((lane < 16) ? 0 : 8) + (j & 7) + ((j >> 3) << 4);
        const float* W = t ? p2w1 : p1w1;
        W1pk[gid] = f2bf((k < 157) ? W[k * 256 + n] : 0.0f);
    } else if (gid < N_W1PK + N_W2PK) {                  // phi W2: (256,471), KT=8
        const long q = gid - N_W1PK;
        const int t = (int)(q / 122880), p = (int)(q % 122880);
        const int j = p & 15, lane = (p >> 4) & 31, rem = p >> 9;
        const int kt = rem % 8, nt = rem / 8;
        const int n = nt * 16 + (lane & 15);
        const int k = kt * 32 + ((lane < 16) ? 0 : 8) + (j & 7) + ((j >> 3) << 4);
        const float* W = t ? p2w2 : p1w2;
        W2pk[q] = f2bf((n < 471) ? W[k * 471 + n] : 0.0f);
    } else if (gid < N_W1PK + N_W2PK + N_RW1PK) {        // rho W1: (471,256), KT=15
        const long q = gid - (N_W1PK + N_W2PK);
        const int t = (int)(q / 122880), p = (int)(q % 122880);
        const int j = p & 15, lane = (p >> 4) & 31, rem = p >> 9;
        const int kt = rem % 15, nt = rem / 15;
        const int n = nt * 16 + (lane & 15);
        const int k = kt * 32 + ((lane < 16) ? 0 : 8) + (j & 7) + ((j >> 3) << 4);
        const float* W = t ? r2w1 : r1w1;
        RW1pk[q] = f2bf((k < 471) ? W[k * 256 + n] : 0.0f);
    } else if (gid < N_W1PK + N_W2PK + N_RW1PK + N_BIAS) {
        const int b = (int)(gid - (N_W1PK + N_W2PK + N_RW1PK));
        float v;
        if (b < 512) {                                   // b1
            v = (b < 256) ? p1b1[b] : p2b1[b - 256];
        } else if (b < 1472) {                           // b2 padded to 480
            const int c = b - 512, tt = c / 480, i = c % 480;
            v = (i < 471) ? (tt ? p2b2[i] : p1b2[i]) : 0.0f;
        } else if (b < 1984) {                           // rb1
            const int c = b - 1472;
            v = (c < 256) ? r1b1[c] : r2b1[c - 256];
        } else if (b < 2496) {                           // rw2
            const int c = b - 1984;
            v = (c < 256) ? r1w2[c] : r2w2[c - 256];
        } else {                                         // rb2
            v = (b == 2496) ? r1b2[0] : r2b2[0];
        }
        bias[b] = v;
    }
}

// ---------------------------------------------------------------------------
// Fused main kernel: one WG handles 32 batch rows for one tower.
// 256 threads = 8 wave32's; all intermediates LDS-resident.
// Each wave processes (m-tile, n-tile-PAIR) groups: one A-fragment load feeds
// two WMMAs (2 accumulators), halving LDS A-traffic per matrix op.
// ---------------------------------------------------------------------------
__global__ __launch_bounds__(256)
void gnn_fused(const float* __restrict__ obs,
               const float* __restrict__ act,
               const float* __restrict__ nodes,
               const unsigned char* __restrict__ ws,
               float* __restrict__ out)
{
    const int tid  = threadIdx.x;
    const int lane = tid & 31;
    const int wave = tid >> 5;
    const int m0   = blockIdx.x * 32;
    const int t    = blockIdx.y;

    const unsigned short* W1pk  = (const unsigned short*)(ws + OFF_W1PK)  + (long)t * 40960;
    const unsigned short* W2pk  = (const unsigned short*)(ws + OFF_W2PK)  + (long)t * 122880;
    const unsigned short* RW1pk = (const unsigned short*)(ws + OFF_RW1PK) + (long)t * 122880;
    const float* bias = (const float*)(ws + OFF_BIAS);
    const float* b1t  = bias + t * 256;
    const float* b2t  = bias + 512 + t * 480;
    const float* rb1t = bias + 1472 + t * 256;
    const float* rw2t = bias + 1984 + t * 256;
    const float  rb2t = bias[2496 + t];

    __shared__ __align__(16) unsigned short sX[32 * 160];   // input tile, bf16
    __shared__ __align__(16) unsigned short sH[32 * 256];   // hidden tile, bf16
    __shared__ __align__(16) float          sY[32 * 480];   // object-sum, f32
    __shared__ __align__(16) unsigned short sYb[32 * 480];  // object-sum, bf16
    __shared__ float sP[256];                               // GEMV partials

    for (int i = tid; i < 32 * 480; i += 256) sY[i] = 0.0f;
    __syncthreads();

    const int nlo   = lane & 15;
    const int mhalf = (lane < 16) ? 0 : 8;

    for (int o = 0; o < OBJ; ++o) {
        // ---- Phase 1: gather + bf16-convert inputs into LDS -------------
        for (int i = tid; i < 32 * 160; i += 256) {
            const int r = i / 160, k = i - r * 160;
            const int s = m0 + r;
            float v;
            if      (k < 4)   v = act[s * 4 + k];
            else if (k < 14)  v = obs[s * 55 + (k - 4)];
            else if (k < 29)  v = obs[s * 55 + 10 + o * 15 + (k - 14)];
            else if (k < 157) v = nodes[((long)o * BATCH_N + s) * 128 + (k - 29)];
            else              v = 0.0f;
            sX[i] = f2bf(v);
        }
        __syncthreads();

        // ---- Phase 2: H = relu(X @ W1 + b1)   (32x160)@(160x256) --------
        // 2 m-tiles x 8 n-tile-pairs = 16 groups, 2 per wave.
        for (int g = wave; g < 16; g += 8) {
            const int mt = g >> 3, np = g & 7;
            const int nt0 = np * 2, nt1 = nt0 + 1;
            v8f acc0 = {}, acc1 = {};
#pragma unroll
            for (int kt = 0; kt < 5; ++kt) {
                Frag a  = loadA(sX, 160, mt * 16, kt * 32, lane);
                Frag b0 = loadB(W1pk, 5, nt0, kt, lane);
                Frag b1 = loadB(W1pk, 5, nt1, kt, lane);
                acc0 = __builtin_amdgcn_wmma_f32_16x16x32_bf16(
                           false, a.v, false, b0.v, (short)0, acc0, false, false);
                acc1 = __builtin_amdgcn_wmma_f32_16x16x32_bf16(
                           false, a.v, false, b1.v, (short)0, acc1, false, false);
            }
            const int n0 = nt0 * 16 + nlo, n1 = nt1 * 16 + nlo;
            const float bv0 = b1t[n0], bv1 = b1t[n1];
            const int mbase = mt * 16 + mhalf;
#pragma unroll
            for (int r = 0; r < 8; ++r) {
                sH[(mbase + r) * 256 + n0] = f2bf(fmaxf(acc0[r] + bv0, 0.0f));
                sH[(mbase + r) * 256 + n1] = f2bf(fmaxf(acc1[r] + bv1, 0.0f));
            }
        }
        __syncthreads();

        // ---- Phase 3: Y += relu(H @ W2 + b2)  (32x256)@(256x480) --------
        // 2 m-tiles x 15 n-tile-pairs = 30 groups.
        for (int g = wave; g < 30; g += 8) {
            const int mt = g / 15, np = g - mt * 15;
            const int nt0 = np * 2, nt1 = nt0 + 1;
            v8f acc0 = {}, acc1 = {};
#pragma unroll
            for (int kt = 0; kt < 8; ++kt) {
                Frag a  = loadA(sH, 256, mt * 16, kt * 32, lane);
                Frag b0 = loadB(W2pk, 8, nt0, kt, lane);
                Frag b1 = loadB(W2pk, 8, nt1, kt, lane);
                acc0 = __builtin_amdgcn_wmma_f32_16x16x32_bf16(
                           false, a.v, false, b0.v, (short)0, acc0, false, false);
                acc1 = __builtin_amdgcn_wmma_f32_16x16x32_bf16(
                           false, a.v, false, b1.v, (short)0, acc1, false, false);
            }
            const int n0 = nt0 * 16 + nlo, n1 = nt1 * 16 + nlo;
            const float bv0 = b2t[n0], bv1 = b2t[n1];
            const int mbase = mt * 16 + mhalf;
#pragma unroll
            for (int r = 0; r < 8; ++r) {                // group->wave map fixed: no race
                sY[(mbase + r) * 480 + n0] += fmaxf(acc0[r] + bv0, 0.0f);
                sY[(mbase + r) * 480 + n1] += fmaxf(acc1[r] + bv1, 0.0f);
            }
        }
        __syncthreads();
    }

    // ---- Convert Y-sum to bf16 for GEMM3 A-operand ----------------------
    for (int i = tid; i < 32 * 480; i += 256) sYb[i] = f2bf(sY[i]);
    __syncthreads();

    // ---- Phase 4: R = relu(Y @ RW1 + rb1)  (32x480)@(480x256) -----------
    for (int g = wave; g < 16; g += 8) {
        const int mt = g >> 3, np = g & 7;
        const int nt0 = np * 2, nt1 = nt0 + 1;
        v8f acc0 = {}, acc1 = {};
#pragma unroll
        for (int kt = 0; kt < 15; ++kt) {
            Frag a  = loadA(sYb, 480, mt * 16, kt * 32, lane);
            Frag b0 = loadB(RW1pk, 15, nt0, kt, lane);
            Frag b1 = loadB(RW1pk, 15, nt1, kt, lane);
            acc0 = __builtin_amdgcn_wmma_f32_16x16x32_bf16(
                       false, a.v, false, b0.v, (short)0, acc0, false, false);
            acc1 = __builtin_amdgcn_wmma_f32_16x16x32_bf16(
                       false, a.v, false, b1.v, (short)0, acc1, false, false);
        }
        const int n0 = nt0 * 16 + nlo, n1 = nt1 * 16 + nlo;
        const float bv0 = rb1t[n0], bv1 = rb1t[n1];
        const int mbase = mt * 16 + mhalf;
#pragma unroll
        for (int r = 0; r < 8; ++r) {
            sH[(mbase + r) * 256 + n0] = f2bf(fmaxf(acc0[r] + bv0, 0.0f));
            sH[(mbase + r) * 256 + n1] = f2bf(fmaxf(acc1[r] + bv1, 0.0f));
        }
    }
    __syncthreads();

    // ---- Phase 5: q = R @ rw2 + rb2  (8 threads per row + LDS reduce) ---
    {
        const int row = tid >> 3, c = tid & 7;          // 32 rows x 8 slices
        float p = 0.0f;
        const int nbase = c * 32;
        for (int n = nbase; n < nbase + 32; ++n)
            p += bf2f(sH[row * 256 + n]) * rw2t[n];
        sP[tid] = p;
    }
    __syncthreads();
    if (tid < 32) {
        float q = rb2t;
#pragma unroll
        for (int j = 0; j < 8; ++j) q += sP[tid * 8 + j];
        out[(long)t * BATCH_N + (m0 + tid)] = q;
    }
}

// ---------------------------------------------------------------------------
extern "C" void kernel_launch(void* const* d_in, const int* in_sizes, int n_in,
                              void* d_out, int out_size, void* d_ws, size_t ws_size,
                              hipStream_t stream)
{
    const float* obs  = (const float*)d_in[0];
    const float* act  = (const float*)d_in[1];
    const float* nds  = (const float*)d_in[2];
    const float* p1w1 = (const float*)d_in[3];
    const float* p1b1 = (const float*)d_in[4];
    const float* p1w2 = (const float*)d_in[5];
    const float* p1b2 = (const float*)d_in[6];
    const float* p2w1 = (const float*)d_in[7];
    const float* p2b1 = (const float*)d_in[8];
    const float* p2w2 = (const float*)d_in[9];
    const float* p2b2 = (const float*)d_in[10];
    const float* r1w1 = (const float*)d_in[11];
    const float* r1b1 = (const float*)d_in[12];
    const float* r1w2 = (const float*)d_in[13];
    const float* r1b2 = (const float*)d_in[14];
    const float* r2w1 = (const float*)d_in[15];
    const float* r2b1 = (const float*)d_in[16];
    const float* r2w2 = (const float*)d_in[17];
    const float* r2b2 = (const float*)d_in[18];

    const long totalp = N_W1PK + N_W2PK + N_RW1PK + N_BIAS;
    gnn_prep<<<dim3((unsigned)((totalp + 255) / 256)), dim3(256), 0, stream>>>(
        p1w1, p2w1, p1w2, p2w2, r1w1, r2w1,
        p1b1, p2b1, p1b2, p2b2, r1b1, r2b1, r1w2, r2w2, r1b2, r2b2,
        (unsigned char*)d_ws);

    dim3 grid(BATCH_N / 32, 2);
    gnn_fused<<<grid, dim3(256), 0, stream>>>(
        obs, act, nds, (const unsigned char*)d_ws, (float*)d_out);
}
